// ReluCoarsen_28664611733899
// MI455X (gfx1250) — matrix operand order repaired and checked by reading, not compiled, not run on hardware
//
#include <hip/hip_runtime.h>

typedef __attribute__((ext_vector_type(16))) _Float16 v16h;
typedef __attribute__((ext_vector_type(8)))  _Float16 v8h;
typedef __attribute__((ext_vector_type(2)))  _Float16 v2h;
typedef __attribute__((ext_vector_type(8)))  float    v8f;

#define C_IN_   64
#define C_OUT_  64
#define K_NBR   9
#define K_TOT   (K_NBR * C_IN_)   // 576
#define LDS_STRIDE 584            // 576 + 8 halfs pad -> 1168B row = 73 x 16B (odd -> conflict-free b128)
#define M_TILE  64
#define NWAVES  16
#define BLOCK   (NWAVES * 32)
#define SW_CHUNKS ((C_OUT_ * LDS_STRIDE) / 8)   // 4672 x 16B chunks

// ---- prep: W[576][64] f32 (row-major K x N) -> Wt[64][584] f16 (N-major, padded) ----
__global__ void prep_wT_kernel(const float* __restrict__ W, _Float16* __restrict__ wT) {
    int t = blockIdx.x * blockDim.x + threadIdx.x;
    if (t >= C_OUT_ * LDS_STRIDE) return;
    int n = t / LDS_STRIDE;
    int k = t - n * LDS_STRIDE;
    float v = (k < K_TOT) ? W[k * C_OUT_ + n] : 0.0f;
    wT[t] = (_Float16)v;
}

__global__ __launch_bounds__(BLOCK)
void relu_coarsen_wmma_kernel(const float* __restrict__ lv,
                              const long long* __restrict__ nbr,
                              const _Float16* __restrict__ wT,
                              const float* __restrict__ bias,
                              float* __restrict__ out,
                              int n_coarse) {
    __shared__ __align__(16) _Float16 sA[M_TILE * LDS_STRIDE]; // 74,752 B
    __shared__ __align__(16) _Float16 sW[C_OUT_ * LDS_STRIDE]; // 74,752 B

    const int tid   = threadIdx.x;
    const int lane  = tid & 31;
    const int wave  = tid >> 5;        // 0..15
    const int row0  = blockIdx.x * M_TILE;

    // ---- stage W^T (f16) into LDS with gfx1250 async global->LDS DMA (ASYNCcnt) ----
    {
        typedef __attribute__((address_space(3))) _Float16 lds_f16;
        unsigned swBase = (unsigned)(uintptr_t)(lds_f16*)sW;   // LDS byte offset of sW
        #pragma unroll 2
        for (int i = tid; i < SW_CHUNKS; i += BLOCK) {
            unsigned    ldsb = swBase + (unsigned)i * 16u;
            const void* gptr = (const void*)((const char*)wT + (size_t)i * 16u);
            asm volatile("global_load_async_to_lds_b128 %0, %1, off"
                         :: "v"(ldsb), "v"(gptr)
                         : "memory");
        }
    }

    // ---- gather + relu + f16 convert into A tile [M_TILE][576] (overlaps with async DMA) ----
    // one wave per (m, neighbor j) fine row: 64*9 = 576 rows; lane loads 2 channels (coalesced 256B/row)
    for (int r = wave; r < M_TILE * K_NBR; r += NWAVES) {
        int m = r / K_NBR;
        int j = r - m * K_NBR;
        long long idx = -1;
        int g = row0 + m;
        if (g < n_coarse) idx = nbr[(long long)g * K_NBR + j];
        v2h h2;
        h2.x = (_Float16)0.0f;
        h2.y = (_Float16)0.0f;
        if (idx >= 0) {
            const float2* __restrict__ p = (const float2*)(lv + (long long)idx * C_IN_);
            float2 v = p[lane];
            float x0 = v.x > 0.0f ? v.x : 0.0f;
            float y0 = v.y > 0.0f ? v.y : 0.0f;
            h2.x = (_Float16)x0;
            h2.y = (_Float16)y0;
        }
        *(v2h*)&sA[m * LDS_STRIDE + j * C_IN_ + lane * 2] = h2;
    }

    // drain async DMA into sW, then make both tiles visible to all waves
    asm volatile("s_wait_asynccnt 0x0" ::: "memory");
    __syncthreads();

    // ---- WMMA compute: wave -> (row-tile rt 0..3, col-tile nt 0..3), 18 K-steps of 32 ----
    const int rt    = wave >> 2;
    const int nt    = wave & 3;
    const int half  = lane >> 4;   // 0: lanes 0-15, 1: lanes 16-31
    const int lan16 = lane & 15;

    // A frag (16-bit A 16x32 layout): lanes0-15 halfs[0..7]=K+0..7, [8..15]=K+16..23
    //                                 lanes16-31 halfs[0..7]=K+8..15, [8..15]=K+24..31
    const _Float16* aBase = &sA[(rt * 16 + lan16) * LDS_STRIDE + half * 8];
    // B frag (32x16, lane = column): lanes0-15 halfs[0..7]=K+0..7, [8..15]=K+16..23; lanes16-31 halfs[0..7]=K+8..15, [8..15]=K+24..31
    const _Float16* bBase = &sW[(nt * 16 + lan16) * LDS_STRIDE + half * 8];

    // two accumulators: breaks the WMMA->WMMA RAW chain (dep distance 2)
    v8f acc0 = {};
    v8f acc1 = {};
    #pragma unroll 3
    for (int ks = 0; ks < K_TOT / 32; ks += 2) {
        int kb0 = ks * 32;
        int kb1 = kb0 + 32;
        v8h a0lo = *(const v8h*)(aBase + kb0);
        v8h a0hi = *(const v8h*)(aBase + kb0 + 16);
        v8h b0lo = *(const v8h*)(bBase + kb0);
        v8h b0hi = *(const v8h*)(bBase + kb0 + 16);
        v8h a1lo = *(const v8h*)(aBase + kb1);
        v8h a1hi = *(const v8h*)(aBase + kb1 + 16);
        v8h b1lo = *(const v8h*)(bBase + kb1);
        v8h b1hi = *(const v8h*)(bBase + kb1 + 16);
        v16h a0, b0, a1, b1;
        #pragma unroll
        for (int i = 0; i < 8; ++i) {
            a0[i] = a0lo[i]; a0[i + 8] = a0hi[i];
            b0[i] = b0lo[i]; b0[i + 8] = b0hi[i];
            a1[i] = a1lo[i]; a1[i + 8] = a1hi[i];
            b1[i] = b1lo[i]; b1[i + 8] = b1hi[i];
        }
        acc0 = __builtin_amdgcn_wmma_f32_16x16x32_f16(
                   false, a0, false, b0, (short)0, acc0, false, false);
        acc1 = __builtin_amdgcn_wmma_f32_16x16x32_f16(
                   false, a1, false, b1, (short)0, acc1, false, false);
    }

    // ---- bias + store: C/D layout: VGPR r, lane l -> M = r + 8*(l>=16), N = l%16 ----
    int nn = nt * 16 + lan16;
    float bb = bias[nn];
    int rowBase = row0 + rt * 16 + half * 8;
    float* op = out + (long long)rowBase * C_OUT_ + nn;

    if (row0 + M_TILE <= n_coarse) {        // block-uniform fast path: no per-lane guards
        #pragma unroll
        for (int rr = 0; rr < 8; ++rr)
            op[(long long)rr * C_OUT_] = acc0[rr] + acc1[rr] + bb;
    } else {
        #pragma unroll
        for (int rr = 0; rr < 8; ++rr) {
            if (rowBase + rr < n_coarse)
                op[(long long)rr * C_OUT_] = acc0[rr] + acc1[rr] + bb;
        }
    }
}

extern "C" void kernel_launch(void* const* d_in, const int* in_sizes, int n_in,
                              void* d_out, int out_size, void* d_ws, size_t ws_size,
                              hipStream_t stream) {
    const float*     lv  = (const float*)d_in[0];
    const long long* nbr = (const long long*)d_in[1];   // jax int64
    const float*     W   = (const float*)d_in[2];
    const float*     b   = (const float*)d_in[3];
    float*           out = (float*)d_out;
    _Float16*        wT  = (_Float16*)d_ws;             // 64*584*2 = 74,752 B scratch

    int n_coarse = in_sizes[1] / K_NBR;                 // 250,000

    int prep_elems = C_OUT_ * LDS_STRIDE;
    prep_wT_kernel<<<(prep_elems + 255) / 256, 256, 0, stream>>>(W, wT);

    int nblocks = (n_coarse + M_TILE - 1) / M_TILE;     // 3907
    relu_coarsen_wmma_kernel<<<nblocks, BLOCK, 0, stream>>>(lv, nbr, wT, b, out, n_coarse);
}
